// GridAttention2D_59931973648683
// MI455X (gfx1250) — matrix operand order repaired and checked by reading, not compiled, not run on hardware
//
#include <hip/hip_runtime.h>

typedef __attribute__((ext_vector_type(16))) _Float16 v16h;
typedef __attribute__((ext_vector_type(8)))  _Float16 v8h;
typedef __attribute__((ext_vector_type(8)))  float    v8f;

#define WQKV_ELEMS  (768 * 256)
#define WPROJ_ELEMS (256 * 256)

// ---------- WMMA helper ----------
__device__ __forceinline__ v8f wmma_f32_f16(v16h a, v16h b, v8f c) {
  // D = A(16x32 f16) x B(32x16 f16) + C(16x16 f32)
  return __builtin_amdgcn_wmma_f32_16x16x32_f16(false, a, false, b, (short)0, c,
                                                false, false);
}

// A fragment (16x32, row-major source, stride in halves).
// Lane l: row m = l&15, hi = l>>4. half j in [0,8): K = hi*8 + j ;
// half j in [8,16): K = 16 + hi*8 + (j-8).   (ISA 7.12.2, 16-bit A 16x32)
__device__ __forceinline__ v16h load_a_frag(const _Float16* base, int stride,
                                            int lane) {
  const int m = lane & 15, hi = lane >> 4;
  const _Float16* p = base + m * stride + hi * 8;
  v8h lo = *reinterpret_cast<const v8h*>(p);
  v8h hs = *reinterpret_cast<const v8h*>(p + 16);
  v16h r;
#pragma unroll
  for (int i = 0; i < 8; ++i) { r[i] = lo[i]; r[i + 8] = hs[i]; }
  return r;
}

// B fragment (32x16). B[k][n] = S[n][k] where S is a row-major source
// (e.g. W for X@W^T). Lane l: col n = l&15; half j in [0,8): K = (l>>4)*8 + j;
// half j in [8,16): K = 16 + (l>>4)*8 + (j-8).
__device__ __forceinline__ v16h load_b_frag(const _Float16* base, int stride,
                                            int lane) {
  const int n = lane & 15, hi = lane >> 4;
  const _Float16* p = base + n * stride + hi * 8;
  v8h lo = *reinterpret_cast<const v8h*>(p);
  v8h hs = *reinterpret_cast<const v8h*>(p + 16);
  v16h r;
#pragma unroll
  for (int i = 0; i < 8; ++i) { r[i] = lo[i]; r[i + 8] = hs[i]; }
  return r;
}

// ---------- weight fp32 -> f16 conversion (runs once per launch) ----------
__global__ void ga_convert_weights(const float* __restrict__ wqkv,
                                   const float* __restrict__ wproj,
                                   _Float16* __restrict__ out) {
  int i = blockIdx.x * 256 + threadIdx.x;
  if (i < WQKV_ELEMS)
    out[i] = (_Float16)wqkv[i];
  else if (i < WQKV_ELEMS + WPROJ_ELEMS)
    out[i] = (_Float16)wproj[i - WQKV_ELEMS];
}

// ---------- fused grid-attention kernel: one block per window ----------
// Window wb: b = wb/64, hp = (wb/8)%8, wp = wb%8.
// Token n in [0,64): gh = n>>3, gw = n&7 ; pixel h = gh*8+hp, w = gw*8+wp.
__global__ __launch_bounds__(256)
void ga_main(const float* __restrict__ x, const _Float16* __restrict__ wqkv_h,
             const float* __restrict__ bqkv, const _Float16* __restrict__ wproj_h,
             const float* __restrict__ bproj, float* __restrict__ out) {
  // LDS (total ~152 KB -> 2 blocks / WGP out of 320 KB)
  __shared__ __align__(16) _Float16 Xs[64 * 264];        // X tile, reused as O
  __shared__ __align__(16) _Float16 QKs[64 * 520];       // Q cols 0..255, K cols 256..511
  __shared__ __align__(16) _Float16 Vts[256 * 72];       // V transposed: [d][token]
  __shared__ __align__(16) _Float16 Ps[8 * 16 * 72];     // per-wave P staging (16 rows)

  const int tid  = threadIdx.x;
  const int lane = tid & 31;
  const int wv   = tid >> 5;          // wave id 0..7 == head id
  const int nn   = lane & 15;
  const int hi   = lane >> 4;

  const int wb = blockIdx.x;
  const int b  = wb >> 6;
  const int hp = (wb >> 3) & 7;
  const int wp = wb & 7;

  // ---- Stage 0: gather window [64 tokens x 256 ch] fp32 -> f16 LDS ----
  for (int i = tid; i < 64 * 64; i += 256) {   // 64 rows x 64 float4
    const int row = i >> 6;
    const int c4  = i & 63;
    const int h   = (row & 0x38) + hp;          // (row>>3)*8 + hp
    const int w   = ((row & 7) << 3) + wp;
    const float4 v = *reinterpret_cast<const float4*>(
        x + ((size_t)((b * 64 + h) * 64 + w)) * 256 + c4 * 4);
    _Float16* d = &Xs[row * 264 + c4 * 4];
    d[0] = (_Float16)v.x; d[1] = (_Float16)v.y;
    d[2] = (_Float16)v.z; d[3] = (_Float16)v.w;
  }
  __syncthreads();

  // ---- Stage 1: QKV = X @ Wqkv^T + b  (M=64, N=768, K=256) ----
  // 48 N-tiles, 6 per wave. Q/K stored row-major, V stored transposed.
  for (int t = 0; t < 6; ++t) {
    const int nt = wv * 6 + t;
    const int n0 = nt * 16;
    const float bias = bqkv[n0 + nn];
    v8f acc[4];
#pragma unroll
    for (int mt = 0; mt < 4; ++mt)
#pragma unroll
      for (int r = 0; r < 8; ++r) acc[mt][r] = bias;

#pragma unroll
    for (int kt = 0; kt < 8; ++kt) {
      const v16h bf = load_b_frag(wqkv_h + n0 * 256 + kt * 32, 256, lane);
#pragma unroll
      for (int mt = 0; mt < 4; ++mt) {
        const v16h af = load_a_frag(&Xs[mt * 16 * 264 + kt * 32], 264, lane);
        acc[mt] = wmma_f32_f16(af, bf, acc[mt]);
      }
    }
    if (n0 < 512) {          // Q or K -> row-major
#pragma unroll
      for (int mt = 0; mt < 4; ++mt)
#pragma unroll
        for (int r = 0; r < 8; ++r)
          QKs[(mt * 16 + r + 8 * hi) * 520 + n0 + nn] = (_Float16)acc[mt][r];
    } else {                 // V -> transposed [d][token]
#pragma unroll
      for (int mt = 0; mt < 4; ++mt)
#pragma unroll
        for (int r = 0; r < 8; ++r)
          Vts[(n0 - 512 + nn) * 72 + mt * 16 + r + 8 * hi] = (_Float16)acc[mt][r];
    }
  }
  __syncthreads();           // QKV complete; Xs now dead (reused as O)

  // ---- Stage 2: attention, wave wv handles head wv ----
  const int   h0    = wv * 32;
  const float scale = 0.17677669529663687f;   // 1/sqrt(32)
  _Float16*   Ph    = &Ps[wv * 16 * 72];

  for (int mt = 0; mt < 4; ++mt) {
    // S = Q @ K^T   (16 x 64, K-dim = 32 -> single k-tile)
    v8f st[4];
#pragma unroll
    for (int ntk = 0; ntk < 4; ++ntk)
#pragma unroll
      for (int r = 0; r < 8; ++r) st[ntk][r] = 0.0f;

    const v16h aq = load_a_frag(&QKs[mt * 16 * 520 + h0], 520, lane);
#pragma unroll
    for (int ntk = 0; ntk < 4; ++ntk) {
      const v16h bk = load_b_frag(&QKs[ntk * 16 * 520 + 256 + h0], 520, lane);
      st[ntk] = wmma_f32_f16(aq, bk, st[ntk]);
    }

    // softmax over keys: rows live in (r, hi), cols across 4 tiles x 16 lanes
#pragma unroll
    for (int r = 0; r < 8; ++r) {
      float m = fmaxf(fmaxf(st[0][r], st[1][r]), fmaxf(st[2][r], st[3][r]));
#pragma unroll
      for (int off = 1; off < 16; off <<= 1) m = fmaxf(m, __shfl_xor(m, off, 32));
      float p[4], s = 0.0f;
#pragma unroll
      for (int ntk = 0; ntk < 4; ++ntk) {
        p[ntk] = __expf((st[ntk][r] - m) * scale);
        s += p[ntk];
      }
#pragma unroll
      for (int off = 1; off < 16; off <<= 1) s += __shfl_xor(s, off, 32);
      const float inv = 1.0f / s;
#pragma unroll
      for (int ntk = 0; ntk < 4; ++ntk)
        Ph[(r + 8 * hi) * 72 + ntk * 16 + nn] = (_Float16)(p[ntk] * inv);
    }

    // O(mt) = P @ V_head   (16 x 32, K-dim = 64 -> 2 k-tiles)
#pragma unroll
    for (int nto = 0; nto < 2; ++nto) {
      v8f oc;
#pragma unroll
      for (int r = 0; r < 8; ++r) oc[r] = 0.0f;
#pragma unroll
      for (int kt = 0; kt < 2; ++kt) {
        const v16h ap = load_a_frag(&Ph[kt * 32], 72, lane);
        const v16h bv = load_b_frag(&Vts[(h0 + nto * 16) * 72 + kt * 32], 72, lane);
        oc = wmma_f32_f16(ap, bv, oc);
      }
#pragma unroll
      for (int r = 0; r < 8; ++r)
        Xs[(mt * 16 + r + 8 * hi) * 264 + h0 + nto * 16 + nn] = (_Float16)oc[r];
    }
  }
  __syncthreads();           // attention output (in Xs) complete

  // ---- Stage 3: Y = O @ Wproj^T + b, scatter to [B,H,W,C] ----
  for (int t = 0; t < 2; ++t) {
    const int nt = wv * 2 + t;
    const int n0 = nt * 16;
    const float bias = bproj[n0 + nn];
    v8f acc[4];
#pragma unroll
    for (int mt = 0; mt < 4; ++mt)
#pragma unroll
      for (int r = 0; r < 8; ++r) acc[mt][r] = bias;

#pragma unroll
    for (int kt = 0; kt < 8; ++kt) {
      const v16h bf = load_b_frag(wproj_h + n0 * 256 + kt * 32, 256, lane);
#pragma unroll
      for (int mt = 0; mt < 4; ++mt) {
        const v16h af = load_a_frag(&Xs[mt * 16 * 264 + kt * 32], 264, lane);
        acc[mt] = wmma_f32_f16(af, bf, acc[mt]);
      }
    }
#pragma unroll
    for (int mt = 0; mt < 4; ++mt) {
#pragma unroll
      for (int r = 0; r < 8; ++r) {
        const int tok = mt * 16 + r + 8 * hi;
        const int h   = (tok & 0x38) + hp;
        const int w   = ((tok & 7) << 3) + wp;
        out[((size_t)((b * 64 + h) * 64 + w)) * 256 + n0 + nn] = acc[mt][r];
      }
    }
  }
}

extern "C" void kernel_launch(void* const* d_in, const int* in_sizes, int n_in,
                              void* d_out, int out_size, void* d_ws, size_t ws_size,
                              hipStream_t stream) {
  const float* x     = (const float*)d_in[0];
  const float* wqkv  = (const float*)d_in[1];
  const float* bqkv  = (const float*)d_in[2];
  const float* wproj = (const float*)d_in[3];
  const float* bproj = (const float*)d_in[4];
  float*       out   = (float*)d_out;

  _Float16* wqkv_h  = (_Float16*)d_ws;
  _Float16* wproj_h = wqkv_h + WQKV_ELEMS;

  ga_convert_weights<<<(WQKV_ELEMS + WPROJ_ELEMS) / 256, 256, 0, stream>>>(
      wqkv, wproj, wqkv_h);
  ga_main<<<2048, 256, 0, stream>>>(x, wqkv_h, bqkv, wproj_h, bproj, out);
}
